// MultiheadedMixtureOfExpertsModel_14345190768798
// MI455X (gfx1250) — matrix operand order repaired and checked by reading, not compiled, not run on hardware
//
#include <hip/hip_runtime.h>
#include <hip/hip_bf16.h>
#include <math.h>

// ---------------------------------------------------------------------------
// MoE model, algebraically collapsed:
//   out = softplus(softplus(x@Weff + beff) @ W2 + b2) @ Wout + bout
//   Weff = (Σ_e w[h,e] W_exp[e]) interleaved -> @ W1   (512 x 256)
// Main GEMMs run as bf16 hi/lo 3-term WMMA (near-fp32 precision).
// ---------------------------------------------------------------------------

typedef __attribute__((ext_vector_type(16))) __bf16 v16bf;
typedef __attribute__((ext_vector_type(8)))  __bf16 v8bf;
typedef __attribute__((ext_vector_type(8)))  float  v8f;

#define N_ROWS   16384
#define IN_DIM   512
#define FEAT     256
#define NHEAD    8
#define NEXP     8
#define HID      256      // 32*H
#define CBIG     2048     // FEAT*H

__device__ __forceinline__ void splitf(float v, __bf16& hi, __bf16& lo) {
    hi = (__bf16)v;
    lo = (__bf16)(v - (float)hi);
}

__device__ __forceinline__ float softplus_f(float v) {
    return v > 15.0f ? v : log1pf(expf(v));
}

__device__ __forceinline__ v8f wmma3(v16bf ah, v16bf al, v16bf bh, v16bf bl, v8f c) {
    c = __builtin_amdgcn_wmma_f32_16x16x32_bf16(false, ah, false, bh, (short)0, c, false, false);
    c = __builtin_amdgcn_wmma_f32_16x16x32_bf16(false, ah, false, bl, (short)0, c, false, false);
    c = __builtin_amdgcn_wmma_f32_16x16x32_bf16(false, al, false, bh, (short)0, c, false, false);
    return c;
}

// --------------------------------------------------------------------------
// Kernel 1: gating. top-2 per head, softmax over selected, dense combine
// weights w[h][e]; bbig[f*8+h]; orthogonality loss -> out[16384].
// --------------------------------------------------------------------------
__global__ void gate_kernel(const float* __restrict__ scaling,
                            const float* __restrict__ b_exp,
                            float* __restrict__ wcomb,
                            float* __restrict__ bbig,
                            float* __restrict__ lossOut) {
    __shared__ float wsm[NHEAD][NEXP];
    int tid = threadIdx.x;
    if (tid < NHEAD * NEXP) ((float*)wsm)[tid] = 0.0f;
    __syncthreads();
    if (tid < NHEAD) {
        int h = tid;
        float v0 = -1e30f, v1 = -1e30f; int i0 = 0, i1 = 0;
        for (int e = 0; e < NEXP; ++e) {
            float v = scaling[h * NEXP + e];
            if (v > v0)      { v1 = v0; i1 = i0; v0 = v; i0 = e; }
            else if (v > v1) { v1 = v;  i1 = e; }
        }
        float e1  = expf(v1 - v0);
        float inv = 1.0f / (1.0f + e1);
        wsm[h][i0] = inv;        // p0
        wsm[h][i1] = e1 * inv;   // p1
    }
    __syncthreads();
    if (tid < NHEAD * NEXP) wcomb[tid] = ((float*)wsm)[tid];
    // bbig[c], c = f*8 + h  (2048 values, 8 per thread)
    for (int q = 0; q < 8; ++q) {
        int c = tid * 8 + q;
        int h = c & 7, f = c >> 3;
        float s = 0.0f;
        #pragma unroll
        for (int e = 0; e < NEXP; ++e) s += wsm[h][e] * b_exp[e * FEAT + f];
        bbig[c] = s;
    }
    if (tid == 0) {
        float loss = 0.0f;
        for (int a = 0; a < NHEAD; ++a)
            for (int b = 0; b < NHEAD; ++b) {
                float g = 0.0f;
                for (int e = 0; e < NEXP; ++e) g += wsm[a][e] * wsm[b][e];
                if (a == b) g -= 1.0f;
                loss += g * g;
            }
        *lossOut = loss;
    }
}

// --------------------------------------------------------------------------
// Kernel 2: Wbig[i][f*8+h] = sum_e w[h,e] * W_exp[e,i,f], stored bf16 hi/lo.
// --------------------------------------------------------------------------
__global__ void build_wbig(const float* __restrict__ W_exp,
                           const float* __restrict__ wcomb,
                           __bf16* __restrict__ WbigHi,
                           __bf16* __restrict__ WbigLo) {
    int idx = blockIdx.x * 256 + threadIdx.x;   // < 512*2048
    int i = idx >> 11;
    int c = idx & (CBIG - 1);
    int h = c & 7, f = c >> 3;
    float s = 0.0f;
    #pragma unroll
    for (int e = 0; e < NEXP; ++e)
        s += wcomb[h * NEXP + e] * W_exp[((size_t)e * IN_DIM + i) * FEAT + f];
    __bf16 hh, ll; splitf(s, hh, ll);
    WbigHi[idx] = hh;
    WbigLo[idx] = ll;
}

// --------------------------------------------------------------------------
// Kernel 3: Weff = Wbig(512x2048) @ W1(2048x256), WMMA bf16x3, fp32 out.
// One 16x16 tile per wave; 64 k-chunks of 32.
// --------------------------------------------------------------------------
__global__ void weff_gemm(const __bf16* __restrict__ WbigHi,
                          const __bf16* __restrict__ WbigLo,
                          const float*  __restrict__ W1,
                          float* __restrict__ Weff) {
    int tid = threadIdx.x;
    int lane = tid & 31, w = tid >> 5;
    int waveId = blockIdx.x * 8 + w;         // 0..511
    int tileR = waveId >> 4;                 // 0..31
    int tileC = waveId & 15;                 // 0..15
    int laneM = lane & 15, h2 = lane >> 4;

    v8f acc = {0.f, 0.f, 0.f, 0.f, 0.f, 0.f, 0.f, 0.f};
    const __bf16* arh = WbigHi + (size_t)(tileR * 16 + laneM) * CBIG;
    const __bf16* arl = WbigLo + (size_t)(tileR * 16 + laneM) * CBIG;
    int coln = tileC * 16 + laneM;

    for (int c = 0; c < CBIG / 32; ++c) {
        int k0 = c * 32 + 8 * h2;
        v8bf a0h = *(const v8bf*)(arh + k0);
        v8bf a1h = *(const v8bf*)(arh + k0 + 16);
        v8bf a0l = *(const v8bf*)(arl + k0);
        v8bf a1l = *(const v8bf*)(arl + k0 + 16);
        v16bf ahi = __builtin_shufflevector(a0h, a1h, 0,1,2,3,4,5,6,7,8,9,10,11,12,13,14,15);
        v16bf alo = __builtin_shufflevector(a0l, a1l, 0,1,2,3,4,5,6,7,8,9,10,11,12,13,14,15);
        v16bf bhi, blo;
        #pragma unroll
        for (int e = 0; e < 16; ++e) {
            float v = W1[(size_t)(c * 32 + 8 * h2 + (e & 7) + 16 * (e >> 3)) * HID + coln];
            __bf16 hh, ll; splitf(v, hh, ll);
            bhi[e] = hh; blo[e] = ll;
        }
        acc = wmma3(ahi, alo, bhi, blo, acc);
    }
    #pragma unroll
    for (int r = 0; r < 8; ++r)
        Weff[(size_t)(tileR * 16 + 8 * h2 + r) * HID + coln] = acc[r];
}

// --------------------------------------------------------------------------
// Kernel 4: beff[j] = b1[j] + sum_c bbig[c]*W1[c][j]
// --------------------------------------------------------------------------
__global__ void beff_kernel(const float* __restrict__ bbig,
                            const float* __restrict__ W1,
                            const float* __restrict__ b1,
                            float* __restrict__ beff) {
    int j = threadIdx.x;
    float s0 = 0.f, s1 = 0.f, s2 = 0.f, s3 = 0.f;
    for (int c = 0; c < CBIG; c += 4) {
        s0 += bbig[c + 0] * W1[(size_t)(c + 0) * HID + j];
        s1 += bbig[c + 1] * W1[(size_t)(c + 1) * HID + j];
        s2 += bbig[c + 2] * W1[(size_t)(c + 2) * HID + j];
        s3 += bbig[c + 3] * W1[(size_t)(c + 3) * HID + j];
    }
    beff[j] = b1[j] + ((s0 + s1) + (s2 + s3));
}

// --------------------------------------------------------------------------
// Kernel 5: pack a fp32 [K][256] matrix into WMMA B-fragment order, bf16 hi/lo.
// Fragment layout: [c=K/32][ct=16][lane=32][e=16], element K = 32c+8*(lane>>4)+(e&7)+16*(e>>3),
// column n = 16*ct + (lane&15). Consumer loads 32 contiguous bytes per lane.
// --------------------------------------------------------------------------
__global__ void pack_bfrag(const float* __restrict__ src, int K,
                           __bf16* __restrict__ dstHi, __bf16* __restrict__ dstLo) {
    int tid = blockIdx.x * blockDim.x + threadIdx.x;
    int total = (K >> 5) * 16 * 32;
    if (tid >= total) return;
    int lane = tid & 31;
    int ct   = (tid >> 5) & 15;
    int c    = tid >> 9;
    int h2 = lane >> 4, laneM = lane & 15;
    int col = ct * 16 + laneM;
    size_t base = (size_t)tid * 16;
    #pragma unroll
    for (int e = 0; e < 16; ++e) {
        float v = src[(size_t)(c * 32 + 8 * h2 + (e & 7) + 16 * (e >> 3)) * HID + col];
        __bf16 hh, ll; splitf(v, hh, ll);
        dstHi[base + e] = hh;
        dstLo[base + e] = ll;
    }
}

// --------------------------------------------------------------------------
// Kernel 6: fused main pipeline. 32 rows per block, 256 threads = 8 waves.
// Wave w: row strip (w&1)*16, column quarter (w>>2... w>>1)*64.
// Layer1: h1 = softplus(x@Weff+beff)  (K=512, 16 chunks, 4 tiles, 3 WMMA each)
// Layer2: h2 = softplus(h1@W2+b2)     (K=256, 8 chunks)
// Out:    out = h2@Wout + bout via LDS f32 atomics.
// --------------------------------------------------------------------------
__global__ void __launch_bounds__(256)
fused_main(const float* __restrict__ x,
           const __bf16* __restrict__ pWeffHi, const __bf16* __restrict__ pWeffLo,
           const float* __restrict__ beff,
           const __bf16* __restrict__ pW2Hi,   const __bf16* __restrict__ pW2Lo,
           const float* __restrict__ b2,
           const float* __restrict__ Wout, const float* __restrict__ bout,
           float* __restrict__ out) {
    __shared__ __bf16 h1hi[32][264];   // pad 8 bf16: row stride 528B -> no bank conflicts
    __shared__ __bf16 h1lo[32][264];
    __shared__ float  outStage[32];

    int tid = threadIdx.x;
    int lane = tid & 31, w = tid >> 5;
    int rowStrip = w & 1;        // 0..1
    int colQ     = w >> 1;       // 0..3 (64 columns each)
    int laneM = lane & 15, h2 = lane >> 4;
    int rowBase = blockIdx.x * 32 + rowStrip * 16;

    if (tid < 32) outStage[tid] = 0.0f;

    // ---------------- Layer 1: x @ Weff + beff -> softplus -> LDS ----------
    v8f acc[4];
    #pragma unroll
    for (int t = 0; t < 4; ++t) {
        float b = beff[colQ * 64 + t * 16 + laneM];
        #pragma unroll
        for (int r = 0; r < 8; ++r) acc[t][r] = b;
    }

    const float* xrow = x + (size_t)(rowBase + laneM) * IN_DIM;
    for (int c = 0; c < IN_DIM / 32; ++c) {            // 16 chunks
        const float* s0 = xrow + c * 32 + 8 * h2;
        const float* s1 = s0 + 16;
        v16bf ahi, alo;
        #pragma unroll
        for (int e = 0; e < 8; ++e) {
            __bf16 hh, ll;
            splitf(s0[e], hh, ll); ahi[e]     = hh; alo[e]     = ll;
            splitf(s1[e], hh, ll); ahi[8 + e] = hh; alo[8 + e] = ll;
        }
        #pragma unroll
        for (int t = 0; t < 4; ++t) {
            int ct = colQ * 4 + t;
            size_t base = ((size_t)(c * 16 + ct) * 32 + lane) * 16;
            v16bf bhi = *(const v16bf*)(pWeffHi + base);
            v16bf blo = *(const v16bf*)(pWeffLo + base);
            acc[t] = wmma3(ahi, alo, bhi, blo, acc[t]);
        }
    }

    #pragma unroll
    for (int t = 0; t < 4; ++t) {
        int col = colQ * 64 + t * 16 + laneM;
        #pragma unroll
        for (int r = 0; r < 8; ++r) {
            int rl = rowStrip * 16 + 8 * h2 + r;
            float s = softplus_f(acc[t][r]);
            __bf16 hh, ll; splitf(s, hh, ll);
            h1hi[rl][col] = hh;
            h1lo[rl][col] = ll;
        }
    }
    __syncthreads();

    // ---------------- Layer 2: h1 @ W2 + b2 -> softplus --------------------
    v8f acc2[4];
    #pragma unroll
    for (int t = 0; t < 4; ++t) {
        float b = b2[colQ * 64 + t * 16 + laneM];
        #pragma unroll
        for (int r = 0; r < 8; ++r) acc2[t][r] = b;
    }
    int arow = rowStrip * 16 + laneM;
    for (int c2 = 0; c2 < HID / 32; ++c2) {            // 8 chunks
        int k0 = c2 * 32 + 8 * h2;
        v8bf a0h = *(const v8bf*)(&h1hi[arow][k0]);
        v8bf a1h = *(const v8bf*)(&h1hi[arow][k0 + 16]);
        v8bf a0l = *(const v8bf*)(&h1lo[arow][k0]);
        v8bf a1l = *(const v8bf*)(&h1lo[arow][k0 + 16]);
        v16bf ahi = __builtin_shufflevector(a0h, a1h, 0,1,2,3,4,5,6,7,8,9,10,11,12,13,14,15);
        v16bf alo = __builtin_shufflevector(a0l, a1l, 0,1,2,3,4,5,6,7,8,9,10,11,12,13,14,15);
        #pragma unroll
        for (int t = 0; t < 4; ++t) {
            int ct = colQ * 4 + t;
            size_t base = ((size_t)(c2 * 16 + ct) * 32 + lane) * 16;
            v16bf bhi = *(const v16bf*)(pW2Hi + base);
            v16bf blo = *(const v16bf*)(pW2Lo + base);
            acc2[t] = wmma3(ahi, alo, bhi, blo, acc2[t]);
        }
    }

    // ---------------- Output: h2 @ Wout + bout -----------------------------
    float wv[4];
    #pragma unroll
    for (int t = 0; t < 4; ++t) wv[t] = Wout[colQ * 64 + t * 16 + laneM];
    #pragma unroll
    for (int r = 0; r < 8; ++r) {
        float p = 0.0f;
        #pragma unroll
        for (int t = 0; t < 4; ++t) p += softplus_f(acc2[t][r]) * wv[t];
        atomicAdd(&outStage[rowStrip * 16 + 8 * h2 + r], p);   // ds_add_f32
    }
    __syncthreads();
    if (tid < 32) out[blockIdx.x * 32 + tid] = outStage[tid] + bout[0];
}

// --------------------------------------------------------------------------
// Workspace layout (bytes). Total ~5.8 MB.
// --------------------------------------------------------------------------
#define OFF_WCOMB     0u          // 64 f32
#define OFF_BBIG      4096u       // 2048 f32
#define OFF_BEFF      16384u      // 256 f32
#define OFF_WEFF      32768u      // 512*256 f32 (512 KB)
#define OFF_WBIG_HI   1048576u    // 512*2048 bf16 (2 MB)
#define OFF_WBIG_LO   3145728u    // 2 MB
#define OFF_PWEFF_HI  5242880u    // 256 KB
#define OFF_PWEFF_LO  5505024u    // 256 KB
#define OFF_PW2_HI    5767168u    // 128 KB
#define OFF_PW2_LO    5898240u    // 128 KB

extern "C" void kernel_launch(void* const* d_in, const int* in_sizes, int n_in,
                              void* d_out, int out_size, void* d_ws, size_t ws_size,
                              hipStream_t stream) {
    (void)in_sizes; (void)n_in; (void)out_size; (void)ws_size;
    const float* x       = (const float*)d_in[0];
    const float* scaling = (const float*)d_in[1];
    const float* W_exp   = (const float*)d_in[2];
    const float* b_exp   = (const float*)d_in[3];
    const float* W1      = (const float*)d_in[4];
    const float* b1      = (const float*)d_in[5];
    const float* W2      = (const float*)d_in[6];
    const float* b2      = (const float*)d_in[7];
    const float* Wout    = (const float*)d_in[8];
    const float* bout    = (const float*)d_in[9];
    float* out = (float*)d_out;

    char* ws = (char*)d_ws;
    float*  wcomb   = (float*)(ws + OFF_WCOMB);
    float*  bbig    = (float*)(ws + OFF_BBIG);
    float*  beff    = (float*)(ws + OFF_BEFF);
    float*  WeffF   = (float*)(ws + OFF_WEFF);
    __bf16* WbigHi  = (__bf16*)(ws + OFF_WBIG_HI);
    __bf16* WbigLo  = (__bf16*)(ws + OFF_WBIG_LO);
    __bf16* pWeffHi = (__bf16*)(ws + OFF_PWEFF_HI);
    __bf16* pWeffLo = (__bf16*)(ws + OFF_PWEFF_LO);
    __bf16* pW2Hi   = (__bf16*)(ws + OFF_PW2_HI);
    __bf16* pW2Lo   = (__bf16*)(ws + OFF_PW2_LO);

    gate_kernel<<<1, 256, 0, stream>>>(scaling, b_exp, wcomb, bbig, out + N_ROWS);
    build_wbig<<<(IN_DIM * CBIG) / 256, 256, 0, stream>>>(W_exp, wcomb, WbigHi, WbigLo);
    weff_gemm<<<64, 256, 0, stream>>>(WbigHi, WbigLo, W1, WeffF);
    beff_kernel<<<1, 256, 0, stream>>>(bbig, W1, b1, beff);
    pack_bfrag<<<32, 256, 0, stream>>>(WeffF, IN_DIM, pWeffHi, pWeffLo);
    pack_bfrag<<<16, 256, 0, stream>>>(W2, HID, pW2Hi, pW2Lo);
    fused_main<<<N_ROWS / 32, 256, 0, stream>>>(x, pWeffHi, pWeffLo, beff,
                                                pW2Hi, pW2Lo, b2, Wout, bout, out);
}
